// GaussianVoxelizer_40467181863368
// MI455X (gfx1250) — matrix operand order repaired and checked by reading, not compiled, not run on hardware
//
#include <hip/hip_runtime.h>
#include <hip/hip_bf16.h>

typedef __attribute__((ext_vector_type(16))) _Float16 v16h;
typedef __attribute__((ext_vector_type(8)))  float    v8f;

// -0.5 * log2(e): folded into the inverse covariance so the inner loop is
// q = sum s_ij d_i d_j ; w = opac * exp2(q) ; mask = (q >= 4 * NHL2E)
#define NHL2E   (-0.72134752044448169f)
#define QTHRESH (-2.8853900817779268f)   // 4.0f * NHL2E

// Raw v_exp_f32: no OCML range fixups. Safe here: any q below the flush
// range is far below QTHRESH, and its weight is force-zeroed by the mask
// select, so flush-to-zero semantics are bit-equivalent for surviving values.
#if defined(__has_builtin) && __has_builtin(__builtin_amdgcn_exp2f)
#define FAST_EXP2(x) __builtin_amdgcn_exp2f(x)
#else
#define FAST_EXP2(x) exp2f(x)
#endif

// ---------------------------------------------------------------------------
// Workspace layout (d_ws):
//   [0, G*16*2)           : B fragments, f16, per-lane WMMA layout
//   [G*32, G*32 + G*48)   : packed params per g (12 floats, 48B):
//                           {mx,my,mz,opac, s00,s01,s02,s11, s12,s22,0,0}
//                           s** = invcov scaled by NHL2E (off-diag also x2)
// ---------------------------------------------------------------------------

__global__ void gv_prep_kernel(const float* __restrict__ means,   // (G,3)
                               const float* __restrict__ opac,    // (G,1)
                               const float* __restrict__ feats,   // (G,16)
                               const float* __restrict__ covs,    // (G,3,3)
                               float* __restrict__ params,        // (G,12)
                               _Float16* __restrict__ Bbuf,       // (G/32,32,16)
                               int G)
{
    int t = threadIdx.x;
    if (t < G) {
        // --- symmetric 3x3 inverse via adjugate, pre-scaled by NHL2E ---
        const float* c = covs + t * 9;
        float c00 = c[0], c01 = c[1], c02 = c[2];
        float c11 = c[4], c12 = c[5], c22 = c[8];
        float a00 = c11 * c22 - c12 * c12;
        float a01 = c02 * c12 - c01 * c22;
        float a02 = c01 * c12 - c02 * c11;
        float a11 = c00 * c22 - c02 * c02;
        float a12 = c01 * c02 - c00 * c12;
        float a22 = c00 * c11 - c01 * c01;
        float det  = c00 * a00 + c01 * a01 + c02 * a02;
        float sd   = NHL2E / det;          // diag scale
        float so   = 2.0f * sd;            // off-diag scale (folds the 2x)
        float* p = params + t * 12;
        p[0] = means[t * 3 + 0];
        p[1] = means[t * 3 + 1];
        p[2] = means[t * 3 + 2];
        p[3] = opac[t];
        p[4] = a00 * sd;  p[5] = a01 * so;  p[6] = a02 * so;  p[7] = a11 * sd;
        p[8] = a12 * so;  p[9] = a22 * sd;  p[10] = 0.f;      p[11] = 0.f;

        // --- pack B fragment (features as f16) in per-lane WMMA-B layout ---
        // thread t = chunk*32 + lane; element e -> K = e (+16 upper half),
        // N = lane & 15.
        int chunk = t >> 5;
        int lane  = t & 31;
        int n     = lane & 15;
        int gbase = chunk * 32 + ((lane & 16) ? 16 : 0);
        _Float16* b = Bbuf + t * 16;
        #pragma unroll
        for (int e = 0; e < 16; ++e) {
            b[e] = (_Float16)feats[(gbase + e) * 16 + n];
        }
    }
}

template <int NCHUNKS>
__global__ void __launch_bounds__(256)
gv_splat_kernel(const float* __restrict__ grid,      // (Nvox,3)
                const float* __restrict__ params,    // (G,12)
                const _Float16* __restrict__ Bbuf,   // (G/32,32,16)
                float* __restrict__ out_dens,        // (Nvox)
                float* __restrict__ out_feat,        // (Nvox,16)
                int n_tiles, int n_chunks_rt)
{
    const int nc = NCHUNKS ? NCHUNKS : n_chunks_rt;
    const int G  = nc * 32;

    // ---- stage all Gaussian params in LDS (once per 8-wave block) ----
    extern __shared__ float sParams[];               // G*12 floats
    for (int i = threadIdx.x; i < G * 12; i += blockDim.x)
        sParams[i] = params[i];
    __syncthreads();

    int wave = (int)((blockIdx.x * blockDim.x + threadIdx.x) >> 5);
    int lane = (int)(threadIdx.x & 31);
    if (wave >= n_tiles) return;           // wave-uniform: EXEC stays all-1s

    int m     = lane & 15;                 // voxel row within tile
    int voxel = wave * 16 + m;
    float gx = grid[voxel * 3 + 0];
    float gy = grid[voxel * 3 + 1];
    float gz = grid[voxel * 3 + 2];

    int koff = (lane & 16) ? 16 : 0;       // A-matrix upper-half K offset

    float cnt_p  = 0.0f;                   // mask count partial (this half)
    float dens_p = 0.0f;                   // density partial (this half)
    v8f acc = {};                          // f32 C/D accumulator (16x16)

    #pragma unroll
    for (int c = 0; c < nc; ++c) {
        // ---- tile-invariant B fragment (pre-packed f16 features) ----
        v16h b = *(const v16h*)(Bbuf + (size_t)(c * 32 + lane) * 16);

        // ---- build A fragment: 16x32 f16 weights for this K-chunk ----
        // element e -> K = e + koff   (contiguous, matches packed B)
        v16h a;
        int gbase = c * 32 + koff;
        #pragma unroll
        for (int e = 0; e < 16; ++e) {
            int g = gbase + e;
            const float4* pv = (const float4*)(sParams + g * 12);
            float4 p0 = pv[0];             // mean.xyz, opac
            float4 p1 = pv[1];             // s00,s01,s02,s11
            float4 p2 = pv[2];             // s12,s22,-,-
            float dx = gx - p0.x;
            float dy = gy - p0.y;
            float dz = gz - p0.z;
            float q = p1.x * dx * dx + p1.w * dy * dy + p2.y * dz * dz
                    + p1.y * dx * dy + p1.z * dx * dz + p2.x * dy * dz;
            // branchless: one raw v_exp_f32, select with v_cndmask
            float w  = p0.w * FAST_EXP2(q);
            bool  in = (q >= QTHRESH);     // == (maha <= 4.0)
            w = in ? w : 0.0f;
            cnt_p  += in ? 1.0f : 0.0f;
            dens_p += w;
            a[e] = (_Float16)w;
        }
        // D = A x B + C  (v_wmma_f32_16x16x32_f16)
        acc = __builtin_amdgcn_wmma_f32_16x16x32_f16(
                  /*neg_a=*/false, a, /*neg_b=*/false, b,
                  /*c_mod=*/(short)0, acc, /*reuse_a=*/false, /*reuse_b=*/false);
    }

    // ---- combine the two half-wave partials (lane m and lane m+16) ----
    float cnt  = cnt_p  + __shfl_xor(cnt_p,  16, 32);
    float dens = dens_p + __shfl_xor(dens_p, 16, 32);
    float invd = 1.0f / fmaxf(cnt, 1e-6f);

    if (lane < 16) out_dens[voxel] = dens * invd;

    // ---- scale & store features: D VGPR r -> row M=r (lo) / M=r+8 (hi) ----
    int half8 = (lane & 16) ? 8 : 0;
    #pragma unroll
    for (int r = 0; r < 8; ++r) {
        float id  = __shfl(invd, r + half8, 32);   // denom of row M=r(+8)
        int   vox = wave * 16 + r + half8;
        out_feat[(size_t)vox * 16 + (lane & 15)] = acc[r] * id;
    }
}

extern "C" void kernel_launch(void* const* d_in, const int* in_sizes, int n_in,
                              void* d_out, int out_size, void* d_ws, size_t ws_size,
                              hipStream_t stream)
{
    const float* grid  = (const float*)d_in[0];   // (N,3)
    const float* means = (const float*)d_in[1];   // (1,G,3)
    const float* opac  = (const float*)d_in[2];   // (1,G,1)
    const float* feats = (const float*)d_in[3];   // (1,G,16)
    const float* covs  = (const float*)d_in[4];   // (1,G,3,3)

    int Nvox = in_sizes[0] / 3;                   // 80000
    int G    = in_sizes[1] / 3;                   // 128

    _Float16* Bbuf   = (_Float16*)d_ws;                        // G*16 halves
    float*    params = (float*)((char*)d_ws + (size_t)G * 32); // G*12 floats

    gv_prep_kernel<<<1, G, 0, stream>>>(means, opac, feats, covs,
                                        params, Bbuf, G);

    float* out_dens = (float*)d_out;
    float* out_feat = out_dens + Nvox;

    int n_tiles  = Nvox / 16;                     // 5000 (Nvox % 16 == 0)
    int n_chunks = G / 32;                        // 4
    int blocks   = (n_tiles + 7) / 8;             // 8 waves (tiles) per block
    size_t smem  = (size_t)G * 12 * sizeof(float);

    if (n_chunks == 4) {
        gv_splat_kernel<4><<<blocks, 256, smem, stream>>>(
            grid, params, Bbuf, out_dens, out_feat, n_tiles, n_chunks);
    } else {
        gv_splat_kernel<0><<<blocks, 256, smem, stream>>>(
            grid, params, Bbuf, out_dens, out_feat, n_tiles, n_chunks);
    }
}
